// MultiHeadAttention_27427661152487
// MI455X (gfx1250) — matrix operand order repaired and checked by reading, not compiled, not run on hardware
//
#include <hip/hip_runtime.h>

// MHA for B=1,S=4096,D=512,H=8,DK=64 on gfx1250 (wave32, WMMA bf16).
// Pipeline: f32->bf16 convert -> fused QKV WMMA GEMM (V stored transposed
// per head) -> flash attention (S^T trick: softmax stats in-lane, P tile is
// directly the A-fragment for P@V) -> dense WMMA GEMM -> f32 out.

typedef __bf16 bf16_t;
typedef __bf16 v16bf __attribute__((ext_vector_type(16)));
typedef __bf16 v8bf  __attribute__((ext_vector_type(8)));
typedef __bf16 v4bf  __attribute__((ext_vector_type(4)));
typedef float  v8f   __attribute__((ext_vector_type(8)));

#define SEQ    4096
#define DMODEL 512
#define NHEAD  8
#define HDK    64
#define L2E    1.4426950408889634f

__device__ __forceinline__ v8f wmma_bf16(v16bf a, v16bf b, v8f c) {
  return __builtin_amdgcn_wmma_f32_16x16x32_bf16(false, a, false, b, (short)0, c,
                                                 false, false);
}

// A-fragment (16x32, row-major source): lane L<16 -> row L, K {0..7,16..23};
// lane L>=16 -> row L-16, K {8..15,24..31}. Two 16B loads per lane.
__device__ __forceinline__ v16bf load_frag_a(const bf16_t* __restrict__ base, int ld) {
  const int lane = threadIdx.x & 31;
  const bf16_t* p = base + (lane & 15) * ld + ((lane >> 4) << 3);
  v8bf lo = *reinterpret_cast<const v8bf*>(p);
  v8bf hi = *reinterpret_cast<const v8bf*>(p + 16);
  v16bf f;
#pragma unroll
  for (int i = 0; i < 8; ++i) { f[i] = lo[i]; f[i + 8] = hi[i]; }
  return f;
}

// B-fragment loaded from B^T (NxK row-major): lane L -> col (L&15),
// K {0..7,16..23} (L<16) / {8..15,24..31} (L>=16). Two 16B loads per lane.
__device__ __forceinline__ v16bf load_frag_b(const bf16_t* __restrict__ base, int ld) {
  const int lane = threadIdx.x & 31;
  const bf16_t* p = base + (lane & 15) * ld + ((lane >> 4) << 3);
  v8bf lo = *reinterpret_cast<const v8bf*>(p);
  v8bf hi = *reinterpret_cast<const v8bf*>(p + 16);
  v16bf f;
#pragma unroll
  for (int i = 0; i < 8; ++i) { f[i] = lo[i]; f[i + 8] = hi[i]; }
  return f;
}

__global__ void __launch_bounds__(256) cvt_f32_to_bf16(
    const float* __restrict__ in, bf16_t* __restrict__ out, int n4) {
  const int i = blockIdx.x * blockDim.x + threadIdx.x;
  if (i >= n4) return;
  const float4 v = reinterpret_cast<const float4*>(in)[i];
  v4bf o;
  o[0] = (bf16_t)v.x; o[1] = (bf16_t)v.y; o[2] = (bf16_t)v.z; o[3] = (bf16_t)v.w;
  reinterpret_cast<v4bf*>(out)[i] = o;
}

// Fused QKV projection: C = X(4096x512) * W^T + b for W in {wq,wk,wv}.
// grid = (32, 12): blockIdx.y selects matrix (y>>2) and 128-col slab (y&3).
// Q,K stored flat bf16 [4096*512] (head slices contiguous). V stored
// transposed per head: Vt[h][d][s].
__global__ void __launch_bounds__(256) qkv_gemm(
    const bf16_t* __restrict__ xb,
    const bf16_t* __restrict__ wqb, const bf16_t* __restrict__ wkb,
    const bf16_t* __restrict__ wvb,
    const float* __restrict__ bq, const float* __restrict__ bk,
    const float* __restrict__ bv,
    bf16_t* __restrict__ Q, bf16_t* __restrict__ Kout, bf16_t* __restrict__ Vt) {
  const int wave = threadIdx.x >> 5;
  const int lane = threadIdx.x & 31;
  const int which = blockIdx.y >> 2;
  const int n0 = (blockIdx.y & 3) * 128 + (wave & 1) * 64;
  const int m0 = blockIdx.x * 128 + (wave >> 1) * 32;
  const bf16_t* w   = (which == 0) ? wqb : (which == 1) ? wkb : wvb;
  const float* bias = (which == 0) ? bq  : (which == 1) ? bk  : bv;

  v8f acc[2][4] = {};
  for (int k0 = 0; k0 < DMODEL; k0 += 32) {
    v16bf a0 = load_frag_a(xb + (m0 +  0) * DMODEL + k0, DMODEL);
    v16bf a1 = load_frag_a(xb + (m0 + 16) * DMODEL + k0, DMODEL);
    v16bf b0 = load_frag_b(w + (n0 +  0) * DMODEL + k0, DMODEL);
    v16bf b1 = load_frag_b(w + (n0 + 16) * DMODEL + k0, DMODEL);
    v16bf b2 = load_frag_b(w + (n0 + 32) * DMODEL + k0, DMODEL);
    v16bf b3 = load_frag_b(w + (n0 + 48) * DMODEL + k0, DMODEL);
    acc[0][0] = wmma_bf16(a0, b0, acc[0][0]);
    acc[0][1] = wmma_bf16(a0, b1, acc[0][1]);
    acc[0][2] = wmma_bf16(a0, b2, acc[0][2]);
    acc[0][3] = wmma_bf16(a0, b3, acc[0][3]);
    acc[1][0] = wmma_bf16(a1, b0, acc[1][0]);
    acc[1][1] = wmma_bf16(a1, b1, acc[1][1]);
    acc[1][2] = wmma_bf16(a1, b2, acc[1][2]);
    acc[1][3] = wmma_bf16(a1, b3, acc[1][3]);
  }

  const int nl = lane & 15;
  const int mb = (lane >> 4) << 3;
#pragma unroll
  for (int i = 0; i < 2; ++i) {
#pragma unroll
    for (int j = 0; j < 4; ++j) {
      const int col = n0 + j * 16 + nl;
      const float bval = bias[col];
#pragma unroll
      for (int r = 0; r < 8; ++r) {
        const int row = m0 + i * 16 + mb + r;
        const float val = acc[i][j][r] + bval;
        if (which == 0) {
          Q[row * DMODEL + col] = (bf16_t)val;
        } else if (which == 1) {
          Kout[row * DMODEL + col] = (bf16_t)val;
        } else {
          const int flat = row * DMODEL + col;     // h*2^18 + s*64 + d
          const int h = flat >> 18;
          const int s = (flat >> 6) & (SEQ - 1);
          const int d = flat & (HDK - 1);
          Vt[h * (HDK * SEQ) + d * SEQ + s] = (bf16_t)val;
        }
      }
    }
  }
}

// Flash attention. grid=(32, 8): 128 q rows per block (8 waves x 16 q), head y.
// S^T = K * Q^T per 16x16 tile -> lanes index q, so softmax stats are in-lane
// + one xor-16 shuffle; the exp'd P tile is exactly the A-fragment for P@V.
__global__ void __launch_bounds__(256) flash_attn(
    const bf16_t* __restrict__ Q, const bf16_t* __restrict__ K,
    const bf16_t* __restrict__ Vt, const int* __restrict__ mask,
    bf16_t* __restrict__ O) {
  const int h = blockIdx.y;
  const int wave = threadIdx.x >> 5;
  const int lane = threadIdx.x & 31;
  const int q0 = blockIdx.x * 128 + wave * 16;

  const bf16_t* Qh = Q + h * (SEQ * HDK);
  const bf16_t* Kh = K + h * (SEQ * HDK);
  const bf16_t* Vh = Vt + h * (HDK * SEQ);

  const v16bf qf0 = load_frag_b(Qh + q0 * HDK + 0, HDK);
  const v16bf qf1 = load_frag_b(Qh + q0 * HDK + 32, HDK);

  v8f o[4] = {};
  float mrun = -__builtin_inff();
  float lrun = 0.f;
  const int qrow = q0 + (lane & 15);
  const int* __restrict__ mrow = mask + qrow * SEQ;
  const int mb = (lane >> 4) << 3;

  for (int k0 = 0; k0 < SEQ; k0 += 32) {
    // S^T tiles: tile0 covers k0..k0+15, tile1 covers k0+16..k0+31
    v16bf ka0 = load_frag_a(Kh + k0 * HDK + 0, HDK);
    v16bf ka1 = load_frag_a(Kh + k0 * HDK + 32, HDK);
    v16bf kb0 = load_frag_a(Kh + (k0 + 16) * HDK + 0, HDK);
    v16bf kb1 = load_frag_a(Kh + (k0 + 16) * HDK + 32, HDK);
    v8f st0 = {}, st1 = {};
    st0 = wmma_bf16(ka0, qf0, st0);
    st0 = wmma_bf16(ka1, qf1, st0);
    st1 = wmma_bf16(kb0, qf0, st1);
    st1 = wmma_bf16(kb1, qf1, st1);

    // mask: lane holds q = lane&15, k = k0 + mb + i (tile0) / +16 (tile1)
    const int4* mp0 = reinterpret_cast<const int4*>(mrow + k0 + mb);
    const int4 m00 = mp0[0], m01 = mp0[1];
    const int4* mp1 = reinterpret_cast<const int4*>(mrow + k0 + 16 + mb);
    const int4 m10 = mp1[0], m11 = mp1[1];
    const float msk0[8] = {(float)m00.x, (float)m00.y, (float)m00.z, (float)m00.w,
                           (float)m01.x, (float)m01.y, (float)m01.z, (float)m01.w};
    const float msk1[8] = {(float)m10.x, (float)m10.y, (float)m10.z, (float)m10.w,
                           (float)m11.x, (float)m11.y, (float)m11.z, (float)m11.w};

    float s0[8], s1[8];
    float smax = -__builtin_inff();
#pragma unroll
    for (int i = 0; i < 8; ++i) {
      s0[i] = st0[i] * 0.125f - 1e9f * msk0[i];
      s1[i] = st1[i] * 0.125f - 1e9f * msk1[i];
      smax = fmaxf(smax, fmaxf(s0[i], s1[i]));
    }
    smax = fmaxf(smax, __shfl_xor(smax, 16));
    const float mnew = fmaxf(mrun, smax);
    const float alpha = exp2f((mrun - mnew) * L2E);

    float p0[8], p1[8];
    float psum = 0.f;
#pragma unroll
    for (int i = 0; i < 8; ++i) {
      p0[i] = exp2f((s0[i] - mnew) * L2E);
      p1[i] = exp2f((s1[i] - mnew) * L2E);
      psum += p0[i] + p1[i];
    }
    psum += __shfl_xor(psum, 16);
    lrun = lrun * alpha + psum;
    mrun = mnew;

    // rescale O accumulators: O-tile row m lives at vgpr i, half (lane>>4)
    float am[8];
#pragma unroll
    for (int i = 0; i < 8; ++i) am[i] = __shfl(alpha, mb + i);
#pragma unroll
    for (int t = 0; t < 4; ++t)
#pragma unroll
      for (int i = 0; i < 8; ++i) o[t][i] *= am[i];

    // P as A-fragment (16q x 32k): pf[i]=tile0 vgpr i, pf[i+8]=tile1 vgpr i
    v16bf pf;
#pragma unroll
    for (int i = 0; i < 8; ++i) { pf[i] = (bf16_t)p0[i]; pf[i + 8] = (bf16_t)p1[i]; }

#pragma unroll
    for (int t = 0; t < 4; ++t) {
      v16bf vf = load_frag_b(Vh + (t * 16) * SEQ + k0, SEQ);
      o[t] = wmma_bf16(pf, vf, o[t]);
    }
  }

  float lm[8];
#pragma unroll
  for (int i = 0; i < 8; ++i) lm[i] = __shfl(lrun, mb + i);
  // output with head-interleave undone: O[s][h*64+d]
#pragma unroll
  for (int t = 0; t < 4; ++t) {
#pragma unroll
    for (int i = 0; i < 8; ++i) {
      const int row = q0 + mb + i;
      const int col = h * HDK + t * 16 + (lane & 15);
      O[row * DMODEL + col] = (bf16_t)(o[t][i] / lm[i]);
    }
  }
}

// Final dense: out = O(4096x512) * Wd^T + b, f32 output. grid=(32,4).
__global__ void __launch_bounds__(256) dense_gemm(
    const bf16_t* __restrict__ A, const bf16_t* __restrict__ Wb,
    const float* __restrict__ bias, float* __restrict__ out) {
  const int wave = threadIdx.x >> 5;
  const int lane = threadIdx.x & 31;
  const int m0 = blockIdx.x * 128 + (wave >> 1) * 32;
  const int n0 = blockIdx.y * 128 + (wave & 1) * 64;

  v8f acc[2][4] = {};
  for (int k0 = 0; k0 < DMODEL; k0 += 32) {
    v16bf a0 = load_frag_a(A + (m0 +  0) * DMODEL + k0, DMODEL);
    v16bf a1 = load_frag_a(A + (m0 + 16) * DMODEL + k0, DMODEL);
    v16bf b0 = load_frag_b(Wb + (n0 +  0) * DMODEL + k0, DMODEL);
    v16bf b1 = load_frag_b(Wb + (n0 + 16) * DMODEL + k0, DMODEL);
    v16bf b2 = load_frag_b(Wb + (n0 + 32) * DMODEL + k0, DMODEL);
    v16bf b3 = load_frag_b(Wb + (n0 + 48) * DMODEL + k0, DMODEL);
    acc[0][0] = wmma_bf16(a0, b0, acc[0][0]);
    acc[0][1] = wmma_bf16(a0, b1, acc[0][1]);
    acc[0][2] = wmma_bf16(a0, b2, acc[0][2]);
    acc[0][3] = wmma_bf16(a0, b3, acc[0][3]);
    acc[1][0] = wmma_bf16(a1, b0, acc[1][0]);
    acc[1][1] = wmma_bf16(a1, b1, acc[1][1]);
    acc[1][2] = wmma_bf16(a1, b2, acc[1][2]);
    acc[1][3] = wmma_bf16(a1, b3, acc[1][3]);
  }

  const int nl = lane & 15;
  const int mb = (lane >> 4) << 3;
#pragma unroll
  for (int i = 0; i < 2; ++i) {
#pragma unroll
    for (int j = 0; j < 4; ++j) {
      const int col = n0 + j * 16 + nl;
      const float bval = bias[col];
#pragma unroll
      for (int r = 0; r < 8; ++r) {
        const int row = m0 + i * 16 + mb + r;
        out[row * DMODEL + col] = acc[i][j][r] + bval;
      }
    }
  }
}

extern "C" void kernel_launch(void* const* d_in, const int* in_sizes, int n_in,
                              void* d_out, int out_size, void* d_ws, size_t ws_size,
                              hipStream_t stream) {
  (void)in_sizes; (void)n_in; (void)out_size; (void)ws_size;
  const float* x    = (const float*)d_in[0];
  const int*   mask = (const int*)d_in[1];
  const float* wq_w = (const float*)d_in[2];
  const float* wq_b = (const float*)d_in[3];
  const float* wk_w = (const float*)d_in[4];
  const float* wk_b = (const float*)d_in[5];
  const float* wv_w = (const float*)d_in[6];
  const float* wv_b = (const float*)d_in[7];
  const float* wd_w = (const float*)d_in[8];
  const float* wd_b = (const float*)d_in[9];

  char* ws = (char*)d_ws;
  const size_t MB = 1024 * 1024;
  bf16_t* xb  = (bf16_t*)(ws + 0);                 // 4 MB  (4096x512 bf16)
  bf16_t* wqb = (bf16_t*)(ws + 4 * MB);            // 512 KB each
  bf16_t* wkb = (bf16_t*)(ws + 4 * MB + 512 * 1024);
  bf16_t* wvb = (bf16_t*)(ws + 5 * MB);
  bf16_t* wdb = (bf16_t*)(ws + 5 * MB + 512 * 1024);
  bf16_t* Qb  = (bf16_t*)(ws + 6 * MB);            // 4 MB flat [4096*512]
  bf16_t* Kb  = (bf16_t*)(ws + 10 * MB);           // 4 MB
  bf16_t* Vtb = (bf16_t*)(ws + 14 * MB);           // 4 MB, [8][64][4096]
  bf16_t* Ob  = (bf16_t*)(ws + 18 * MB);           // 4 MB, [4096][512]

  cvt_f32_to_bf16<<<2048, 256, 0, stream>>>(x, xb, (SEQ * DMODEL) / 4);
  cvt_f32_to_bf16<<<256, 256, 0, stream>>>(wq_w, wqb, (DMODEL * DMODEL) / 4);
  cvt_f32_to_bf16<<<256, 256, 0, stream>>>(wk_w, wkb, (DMODEL * DMODEL) / 4);
  cvt_f32_to_bf16<<<256, 256, 0, stream>>>(wv_w, wvb, (DMODEL * DMODEL) / 4);
  cvt_f32_to_bf16<<<256, 256, 0, stream>>>(wd_w, wdb, (DMODEL * DMODEL) / 4);

  qkv_gemm<<<dim3(32, 12), 256, 0, stream>>>(xb, wqb, wkb, wvb,
                                             wq_b, wk_b, wv_b, Qb, Kb, Vtb);
  flash_attn<<<dim3(32, NHEAD), 256, 0, stream>>>(Qb, Kb, Vtb, mask, Ob);
  dense_gemm<<<dim3(32, 4), 256, 0, stream>>>(Ob, wdb, wd_b, (float*)d_out);
}
